// ProteinMHAttentionTransformer_82154134438752
// MI455X (gfx1250) — matrix operand order repaired and checked by reading, not compiled, not run on hardware
//
#include <hip/hip_runtime.h>
#include <hip/hip_bf16.h>

#define Lc 16384
#define Cc 256
#define Hh 16
#define Wc 15
#define NT 128         // L-tile per block in the GEMM
#define KTOT (Hh * Cc) // 4096

typedef __bf16 bf16;
typedef __attribute__((ext_vector_type(16))) __bf16 v16bf;
typedef __attribute__((ext_vector_type(8)))  __bf16 v8bf;
typedef __attribute__((ext_vector_type(8)))  float  v8f;

// ---------------------------------------------------------------------------
// Kernel 1: w_tr (fp32, [H][C_out][C_in]) -> bf16 A matrix, A[m][h*256+c]
// ---------------------------------------------------------------------------
__global__ __launch_bounds__(256) void wconv_kernel(const float* __restrict__ w_tr,
                                                    unsigned short* __restrict__ A_us) {
    bf16* A = reinterpret_cast<bf16*>(A_us);
    int i = blockIdx.x * 256 + threadIdx.x;          // i = h*65536 + m*256 + c
    if (i < Hh * Cc * Cc) {
        int h = i >> 16;
        int m = (i >> 8) & 255;
        int c = i & 255;
        A[m * KTOT + h * Cc + c] = (bf16)w_tr[i];
    }
}

// ---------------------------------------------------------------------------
// Kernel 2: bias_sum[o] = sum_h b_tr[h][o]
// ---------------------------------------------------------------------------
__global__ __launch_bounds__(256) void bias_kernel(const float* __restrict__ b_tr,
                                                   float* __restrict__ bias_sum) {
    int o = threadIdx.x;
    float s = 0.f;
#pragma unroll
    for (int h = 0; h < Hh; ++h) s += b_tr[h * Cc + o];
    bias_sum[o] = s;
}

// ---------------------------------------------------------------------------
// Kernel 3: scores[h][l] = b_heads[h] + sum_{c,w} x[c][l-7+w] * w_heads[h][c][w]
// grid: (L/256, H), block 256. x rows staged through LDS in chunks of 32.
// ---------------------------------------------------------------------------
__global__ __launch_bounds__(256) void conv_scores_kernel(const float* __restrict__ x,
                                                          const float* __restrict__ wh_g,
                                                          const float* __restrict__ bh,
                                                          float* __restrict__ scores) {
    __shared__ float wh[Cc * Wc];       // 15 KB
    __shared__ float xs[32][272];       // 34 KB (row = 270 valid + pad)
    const int tid = threadIdx.x;
    const int h   = blockIdx.y;
    const int l0  = blockIdx.x * 256;

    for (int i = tid; i < Cc * Wc; i += 256) wh[i] = wh_g[h * Cc * Wc + i];

    float acc = 0.f;
    for (int cb = 0; cb < Cc / 32; ++cb) {
        __syncthreads();   // protect previous chunk's reads
        for (int i = tid; i < 32 * 270; i += 256) {
            int rr = i / 270, j = i % 270;
            int gl = l0 - 7 + j;
            int c  = cb * 32 + rr;
            xs[rr][j] = (gl >= 0 && gl < Lc) ? x[c * Lc + gl] : 0.f;
        }
        __syncthreads();
        for (int rr = 0; rr < 32; ++rr) {
            int c = cb * 32 + rr;
#pragma unroll
            for (int w = 0; w < Wc; ++w) acc += xs[rr][tid + w] * wh[c * Wc + w];
        }
    }
    scores[h * Lc + l0 + tid] = acc + bh[h];
}

// ---------------------------------------------------------------------------
// Kernel 4: per-head softmax over L. grid: H blocks of 256 threads.
// ---------------------------------------------------------------------------
__global__ __launch_bounds__(256) void softmax_kernel(const float* __restrict__ scores,
                                                      float* __restrict__ focus) {
    __shared__ float red[256];
    const int h = blockIdx.x, tid = threadIdx.x;
    const float* s = scores + h * Lc;

    float m = -1e30f;
    for (int l = tid; l < Lc; l += 256) m = fmaxf(m, s[l]);
    red[tid] = m; __syncthreads();
    for (int off = 128; off > 0; off >>= 1) {
        if (tid < off) red[tid] = fmaxf(red[tid], red[tid + off]);
        __syncthreads();
    }
    m = red[0]; __syncthreads();

    float sum = 0.f;
    for (int l = tid; l < Lc; l += 256) sum += __expf(s[l] - m);
    red[tid] = sum; __syncthreads();
    for (int off = 128; off > 0; off >>= 1) {
        if (tid < off) red[tid] += red[tid + off];
        __syncthreads();
    }
    float inv = 1.f / red[0];
    for (int l = tid; l < Lc; l += 256) focus[h * Lc + l] = __expf(s[l] - m) * inv;
}

// ---------------------------------------------------------------------------
// Kernel 5: WMMA GEMM with per-head focus scaling.
// out[o, l0+n] = sum_h f[h,n] * (W_h @ x)[o, n] + bias_sum[o]
// grid: L/NT = 128 blocks, 512 threads (16 waves).
// Wave w: M in [32*(w&7), +32), N-half (w>>3)*64. 2x4 fragments per wave.
// ---------------------------------------------------------------------------
__global__ __launch_bounds__(512) void gemm_kernel(const float* __restrict__ x,
                                                   const unsigned short* __restrict__ A_us,
                                                   const float* __restrict__ focus,
                                                   const float* __restrict__ bias_sum,
                                                   float* __restrict__ out) {
    const bf16* __restrict__ Abf = reinterpret_cast<const bf16*>(A_us);

    __shared__ __align__(16) bf16  xT[NT][Cc + 8];   // [n][c], bf16, ~68 KB
    __shared__ __align__(16) float fl[Hh][NT];       // 8 KB
    __shared__ float bl[Cc];                         // 1 KB

    const int tid  = threadIdx.x;
    const int l0   = blockIdx.x * NT;
    const int lane = tid & 31;
    const int wave = tid >> 5;
    const int m0   = (wave & 7) * 32;
    const int n0   = (wave >> 3) * 64;
    const int ln15 = lane & 15;
    const int half = (lane >> 4) & 1;

    // stage focus tile via gfx1250 async global->LDS copy (16 B per thread,
    // flat layout of fl[][] matches flat global order within this tile)
    {
        int h = tid >> 5;              // (4*tid)>>7
        int n = (4 * tid) & (NT - 1);
        const float* gsrc = focus + h * Lc + l0 + n;
        unsigned lds_off = (unsigned)(size_t)(&fl[0][0]) + 16u * (unsigned)tid;
        asm volatile("global_load_async_to_lds_b128 %0, %1, off"
                     :: "v"(lds_off), "v"(gsrc) : "memory");
    }

    // stage x tile (transposed, bf16) + bias
    for (int idx = tid; idx < Cc * NT; idx += 512) {
        int c = idx >> 7, n = idx & (NT - 1);
        xT[n][c] = (bf16)x[c * Lc + l0 + n];
    }
    if (tid < Cc) bl[tid] = bias_sum[tid];

    asm volatile("s_wait_asynccnt 0x0" ::: "memory");
    __syncthreads();

    v8f acc[2][4];
#pragma unroll
    for (int mf = 0; mf < 2; ++mf)
#pragma unroll
        for (int nf = 0; nf < 4; ++nf)
#pragma unroll
            for (int r = 0; r < 8; ++r) acc[mf][nf][r] = 0.f;

#pragma unroll 1
    for (int h = 0; h < Hh; ++h) {
        v8f y[2][4];
#pragma unroll
        for (int mf = 0; mf < 2; ++mf)
#pragma unroll
            for (int nf = 0; nf < 4; ++nf)
#pragma unroll
                for (int r = 0; r < 8; ++r) y[mf][nf][r] = 0.f;

#pragma unroll 1
        for (int kc = 0; kc < Cc / 32; ++kc) {
            const int k0 = h * Cc + kc * 32;

            // A fragments: 16x32 bf16, ISA layout:
            // lanes 0-15: elems 0-7 <-> K k0+0..7,  elems 8-15 <-> K k0+16..23
            // lanes16-31: elems 0-7 <-> K k0+8..15, elems 8-15 <-> K k0+24..31
            v16bf a[2];
#pragma unroll
            for (int mf = 0; mf < 2; ++mf) {
                int m = m0 + mf * 16 + ln15;
                const bf16* pa = Abf + (size_t)m * KTOT + k0 + half * 8;
                v8bf lo = *(const v8bf*)(pa);
                v8bf hi = *(const v8bf*)(pa + 16);
#pragma unroll
                for (int j = 0; j < 8; ++j) { a[mf][j] = lo[j]; a[mf][j + 8] = hi[j]; }
            }

            // B fragments: 32x16 bf16; same K order as A: elems 0-7 <-> K 8*half+0..7, elems 8-15 <-> K 16+8*half+0..7
            v16bf b[4];
#pragma unroll
            for (int nf = 0; nf < 4; ++nf) {
                int n = n0 + nf * 16 + ln15;
                const bf16* pb = &xT[n][kc * 32 + half * 8];
                v8bf lo = *(const v8bf*)(pb);
                v8bf hi = *(const v8bf*)(pb + 16);
#pragma unroll
                for (int j = 0; j < 8; ++j) { b[nf][j] = lo[j]; b[nf][j + 8] = hi[j]; }
            }

#pragma unroll
            for (int mf = 0; mf < 2; ++mf)
#pragma unroll
                for (int nf = 0; nf < 4; ++nf)
                    y[mf][nf] = __builtin_amdgcn_wmma_f32_16x16x32_bf16(
                        false, a[mf], false, b[nf], (short)0, y[mf][nf], false, false);
        }

        // fold per-head result with focus scale (N fixed per lane in C/D layout)
#pragma unroll
        for (int nf = 0; nf < 4; ++nf) {
            float s = fl[h][n0 + nf * 16 + ln15];
#pragma unroll
            for (int mf = 0; mf < 2; ++mf)
#pragma unroll
                for (int r = 0; r < 8; ++r) acc[mf][nf][r] += s * y[mf][nf][r];
        }
    }

    // epilogue: D layout -> out[o][l0+n] + bias
#pragma unroll
    for (int mf = 0; mf < 2; ++mf)
#pragma unroll
        for (int nf = 0; nf < 4; ++nf) {
            int n = n0 + nf * 16 + ln15;
#pragma unroll
            for (int r = 0; r < 8; ++r) {
                int o = m0 + mf * 16 + r + half * 8;
                out[o * Lc + l0 + n] = acc[mf][nf][r] + bl[o];
            }
        }
}

// ---------------------------------------------------------------------------
extern "C" void kernel_launch(void* const* d_in, const int* in_sizes, int n_in,
                              void* d_out, int out_size, void* d_ws, size_t ws_size,
                              hipStream_t stream) {
    const float* x       = (const float*)d_in[0];   // [1,256,16384]
    const float* w_heads = (const float*)d_in[1];   // [16,256,15]
    const float* b_heads = (const float*)d_in[2];   // [16]
    const float* w_tr    = (const float*)d_in[3];   // [16,256,256]
    const float* b_tr    = (const float*)d_in[4];   // [16,256]
    float* out = (float*)d_out;                     // [1,256,16384]

    // workspace layout (~4.2 MB)
    float* scores   = (float*)d_ws;                 // 16*16384 f32
    float* focus    = scores + Hh * Lc;             // 16*16384 f32
    float* bias_sum = focus + Hh * Lc;              // 256 f32
    unsigned short* A_us = (unsigned short*)(bias_sum + 256); // 256*4096 bf16 (16B aligned)

    wconv_kernel<<<(Hh * Cc * Cc) / 256, 256, 0, stream>>>(w_tr, A_us);
    bias_kernel<<<1, 256, 0, stream>>>(b_tr, bias_sum);
    conv_scores_kernel<<<dim3(Lc / 256, Hh), 256, 0, stream>>>(x, w_heads, b_heads, scores);
    softmax_kernel<<<Hh, 256, 0, stream>>>(scores, focus);
    gemm_kernel<<<Lc / NT, 512, 0, stream>>>(x, A_us, focus, bias_sum, out);
}
